// GAUDenoisingModel_54554674594258
// MI455X (gfx1250) — matrix operand order repaired and checked by reading, not compiled, not run on hardware
//
#include <hip/hip_runtime.h>
#include <hip/hip_bf16.h>

// ---------------------------------------------------------------------------
// Types for CDNA5 WMMA (wave32, v_wmma_f32_16x16x32_bf16)
// ---------------------------------------------------------------------------
typedef __attribute__((ext_vector_type(16))) __bf16         v16bf;
typedef __attribute__((ext_vector_type(8)))  __bf16         bf16x8;
typedef __attribute__((ext_vector_type(8)))  float          v8f;
typedef __attribute__((ext_vector_type(8)))  unsigned short ushort8;
typedef __attribute__((ext_vector_type(4)))  float          f32x4;

__device__ __forceinline__ unsigned short f2bf_u(float f) {
  union { float f; unsigned u; } a; a.f = f;
  unsigned r = a.u + 0x7FFFu + ((a.u >> 16) & 1u);   // round-to-nearest-even
  return (unsigned short)(r >> 16);
}

__device__ __forceinline__ v16bf combine16(bf16x8 lo, bf16x8 hi) {
  return __builtin_shufflevector(lo, hi, 0,1,2,3,4,5,6,7,8,9,10,11,12,13,14,15);
}

// LDS byte address of a __shared__ object: flat LDS aperture keeps the LDS
// offset in addr[31:0] (ISA 10.2 aperture mapping), so truncation is exact.
__device__ __forceinline__ unsigned lds_addr(const void* p) {
  return (unsigned)(size_t)p;
}

// Model constants
#define BB   16
#define LL   256
#define HH   768
#define EE   1536
#define KK   128
#define PD   192
#define ROWS (BB*LL)          // 4096
#define UVQK (2*EE + 2*KK)    // 3328

#define LS   40               // LDS row stride (bf16 elems): 80B, 16B-aligned rows

// ---------------------------------------------------------------------------
// Generic bf16-WMMA GEMM: C(MxN) = A(MxK, bf16 row-major) * B(KxN)
//   BMODE 0: B fp32  row-major [K][N]  (weights, converted to bf16 on load)
//   BMODE 1: B bf16  row-major [K][N]  (activations, e.g. V)
//   BMODE 2: B bf16  N-major   [N][K]  (e.g. K for Q*K^T)
//   EPI   0: C = acc     EPI 1: C += acc (residual)
// Block: 256 thr = 8 waves; tile 128x128; wave tile 32x64; K-step 32.
// Double-buffered LDS, software pipelined:
//   issue(ks+1): async A-tile DMA to LDS (ASYNCcnt) + B slice global->regs
//   compute(ks): 12x ds_load_b128 fragments + 8x v_wmma_f32_16x16x32_bf16
//   commit(ks+1): convert/transpose B regs -> LDS (DScnt)
//   s_wait_asynccnt 0 ; one workgroup barrier per K-step.
// M multiple of 128, K multiple of 32 (true for all call sites).
// ---------------------------------------------------------------------------
template<int BMODE, int EPI>
__global__ void __launch_bounds__(256) gemm_bf16_wmma(
    const unsigned short* __restrict__ A_, const void* __restrict__ B_,
    float* __restrict__ C_,
    int M, int N, int Kd, int lda, int ldb, int ldc,
    long long sA, long long sB, long long sC)
{
  __shared__ unsigned short lA[2][128 * LS];
  __shared__ unsigned short lB[2][128 * LS];

  const int bz = blockIdx.z;
  const unsigned short* A  = A_ + (size_t)bz * sA;
  const float*          Bf = (const float*)B_ + (size_t)bz * sB;
  const unsigned short* Bu = (const unsigned short*)B_ + (size_t)bz * sB;
  float*                C  = C_ + (size_t)bz * sC;

  const int nBase = blockIdx.x * 128;
  const int mBase = blockIdx.y * 128;
  const int tid   = threadIdx.x;
  const int lane  = tid & 31;             // wave32
  const int wave  = tid >> 5;
  const int waveM = (wave & 3) * 32;      // 4 waves along M
  const int waveN = (wave >> 2) * 64;     // 2 waves along N

  const int nSteps = Kd >> 5;

  v8f acc[2][4];
  #pragma unroll
  for (int i = 0; i < 2; ++i)
    #pragma unroll
    for (int j = 0; j < 4; ++j)
      #pragma unroll
      for (int e = 0; e < 8; ++e) acc[i][j][e] = 0.0f;

  f32x4   qreg[4];   // BMODE0 in-flight B slice (fp32)
  ushort8 preg[2];   // BMODE1 in-flight B slice (bf16)

  // ---- issue phase: async A-tile to LDS + B slice loads ----
  auto stage_issue = [&](int ks, int buf) {
    const int k0 = ks << 5;
    {
      const int r = tid >> 1;
      const int c = (tid & 1) << 4;
      const unsigned short* srcA = A + (size_t)(mBase + r) * lda + k0 + c;
      const unsigned dstA = lds_addr(&lA[buf][r * LS + c]);
      asm volatile(
          "global_load_async_to_lds_b128 %0, %2, off\n\t"
          "global_load_async_to_lds_b128 %1, %3, off"
          :: "v"(dstA), "v"(dstA + 16u), "v"(srcA), "v"(srcA + 8)
          : "memory");
    }
    if (BMODE == 2) {
      const int n = tid >> 1;
      const int c = (tid & 1) << 4;
      if (nBase + n < N) {
        const unsigned short* srcB = Bu + (size_t)(nBase + n) * ldb + k0 + c;
        const unsigned dstB = lds_addr(&lB[buf][n * LS + c]);
        asm volatile(
            "global_load_async_to_lds_b128 %0, %2, off\n\t"
            "global_load_async_to_lds_b128 %1, %3, off"
            :: "v"(dstB), "v"(dstB + 16u), "v"(srcB), "v"(srcB + 8)
            : "memory");
      } else {
        #pragma unroll
        for (int e = 0; e < 16; ++e) lB[buf][n * LS + c + e] = 0;
      }
    } else if (BMODE == 0) {
      if (nBase + 128 <= N) {
        const int k  = tid >> 3;            // 0..31
        const int n0 = (tid & 7) << 4;      // 0..112
        const float* src = Bf + (size_t)(k0 + k) * ldb + nBase + n0;
        #pragma unroll
        for (int j = 0; j < 4; ++j) qreg[j] = *reinterpret_cast<const f32x4*>(src + 4 * j);
        if (k0 + 32 < Kd) __builtin_prefetch(src + (size_t)32 * ldb, 0, 0);
      }
    } else if (BMODE == 1) {
      if (nBase + 128 <= N) {
        const int k  = tid >> 3;
        const int n0 = (tid & 7) << 4;
        const unsigned short* src = Bu + (size_t)(k0 + k) * ldb + nBase + n0;
        preg[0] = *reinterpret_cast<const ushort8*>(src);
        preg[1] = *reinterpret_cast<const ushort8*>(src + 8);
      }
    }
  };

  // ---- commit phase: convert/transpose B slice into LDS (after WMMAs) ----
  auto stage_commit = [&](int ks, int buf) {
    const int k0 = ks << 5;
    if (BMODE == 0) {
      const int k  = tid >> 3;
      const int n0 = (tid & 7) << 4;
      if (nBase + 128 <= N) {
        #pragma unroll
        for (int i = 0; i < 16; ++i)
          lB[buf][(n0 + i) * LS + k] = f2bf_u(qreg[i >> 2][i & 3]);
      } else {
        const float* src = Bf + (size_t)(k0 + k) * ldb + nBase + n0;
        #pragma unroll
        for (int i = 0; i < 16; ++i) {
          float v = (nBase + n0 + i < N) ? src[i] : 0.0f;
          lB[buf][(n0 + i) * LS + k] = f2bf_u(v);
        }
      }
    } else if (BMODE == 1) {
      const int k  = tid >> 3;
      const int n0 = (tid & 7) << 4;
      if (nBase + 128 <= N) {
        #pragma unroll
        for (int i = 0; i < 8; ++i) {
          lB[buf][(n0 + i)     * LS + k] = preg[0][i];
          lB[buf][(n0 + 8 + i) * LS + k] = preg[1][i];
        }
      } else {
        const unsigned short* src = Bu + (size_t)(k0 + k) * ldb + nBase + n0;
        #pragma unroll
        for (int i = 0; i < 16; ++i)
          lB[buf][(n0 + i) * LS + k] = (nBase + n0 + i < N) ? src[i] : (unsigned short)0;
      }
    }
  };

  // ---- prologue: fill buffer 0 ----
  stage_issue(0, 0);
  stage_commit(0, 0);
  asm volatile("s_wait_asynccnt 0x0" ::: "memory");
  __syncthreads();

  // ---- main pipelined loop: one barrier per K-step ----
  for (int ks = 0; ks < nSteps; ++ks) {
    const int cur  = ks & 1;
    const bool more = (ks + 1 < nSteps);
    if (more) stage_issue(ks + 1, cur ^ 1);

    // A 16x32 bf16 layout: lanes 0-15 -> M=lane, K{0..7,16..23}; lanes 16-31 -> K{8..15,24..31}
    v16bf af[2];
    #pragma unroll
    for (int mi = 0; mi < 2; ++mi) {
      const int r  = waveM + mi * 16 + (lane & 15);
      const int cc = (lane < 16) ? 0 : 8;
      bf16x8 lo = *reinterpret_cast<const bf16x8*>(&lA[cur][r * LS + cc]);
      bf16x8 hi = *reinterpret_cast<const bf16x8*>(&lA[cur][r * LS + cc + 16]);
      af[mi] = combine16(lo, hi);
    }
    // B 32x16 bf16 layout: lanes 0-15 -> N=lane, K{0..7,16..23}; lanes 16-31 -> K{8..15,24..31}
    v16bf bfr[4];
    #pragma unroll
    for (int ni = 0; ni < 4; ++ni) {
      const int n  = waveN + ni * 16 + (lane & 15);
      const int cc = (lane < 16) ? 0 : 8;
      bf16x8 lo = *reinterpret_cast<const bf16x8*>(&lB[cur][n * LS + cc]);
      bf16x8 hi = *reinterpret_cast<const bf16x8*>(&lB[cur][n * LS + cc + 16]);
      bfr[ni] = combine16(lo, hi);
    }
    #pragma unroll
    for (int mi = 0; mi < 2; ++mi)
      #pragma unroll
      for (int ni = 0; ni < 4; ++ni)
        acc[mi][ni] = __builtin_amdgcn_wmma_f32_16x16x32_bf16(
            false, af[mi], false, bfr[ni], (short)0, acc[mi][ni], false, false);

    if (more) stage_commit(ks + 1, cur ^ 1);
    asm volatile("s_wait_asynccnt 0x0" ::: "memory");
    __syncthreads();
  }

  // ---- epilogue: C/D 16x16 f32 layout (VGPR r: lanes0-15 M=r, lanes16-31 M=r+8)
  #pragma unroll
  for (int mi = 0; mi < 2; ++mi)
    #pragma unroll
    for (int ni = 0; ni < 4; ++ni) {
      const int col = nBase + waveN + ni * 16 + (lane & 15);
      if (col < N) {
        const int rowBase = mBase + waveM + mi * 16 + ((lane < 16) ? 0 : 8);
        #pragma unroll
        for (int r = 0; r < 8; ++r) {
          float* dst = &C[(size_t)(rowBase + r) * ldc + col];
          if (EPI == 1) *dst += acc[mi][ni][r];
          else          *dst  = acc[mi][ni][r];
        }
      }
    }
}

// ---------------------------------------------------------------------------
// Elementwise / reduction helper kernels
// ---------------------------------------------------------------------------

// 2D RoPE table: pe[l][0:32]=sin1 [32:64]=cos1 [64:96]=sin2 [96:128]=cos2
__global__ void pe_kernel(float* __restrict__ pe) {
  const int l = blockIdx.x, i = threadIdx.x;      // grid 256, block 128
  const int p1 = l >> 4, p2 = l & 15;
  const int j  = i & 31;
  const float f   = expf((float)j * (-logf(1000.0f) / 31.0f));
  const float ang = (float)((i < 64) ? p1 : p2) * f;
  pe[l * KK + i] = (((i >> 5) & 1) == 0) ? sinf(ang) : cosf(ang);
}

__global__ void patchify_kernel(const float* __restrict__ x, unsigned short* __restrict__ xp) {
  const int idx = blockIdx.x * 256 + threadIdx.x;     // B*L*PD = 786432
  if (idx >= BB * LL * PD) return;
  const int pd = idx % PD;
  const int l  = (idx / PD) % LL;
  const int b  = idx / (PD * LL);
  const int c  = pd % 3, px = (pd / 3) & 7, py = pd / 24;
  const int gx = l & 15, gy = l >> 4;
  xp[idx] = f2bf_u(x[(((size_t)b * 3 + c) * 128 + (gy * 8 + py)) * 128 + (gx * 8 + px)]);
}

__global__ void add_temb_kernel(float* __restrict__ h, const float* __restrict__ t_emb,
                                const int* __restrict__ t_idx) {
  const int idx = blockIdx.x * 256 + threadIdx.x;     // B*L*H
  if (idx >= ROWS * HH) return;
  const int col = idx % HH;
  const int b   = idx / (LL * HH);
  h[idx] += t_emb[(size_t)t_idx[b] * HH + col];
}

__global__ void rmsnorm_kernel(const float* __restrict__ x, const float* __restrict__ w,
                               unsigned short* __restrict__ out, int ncols) {
  __shared__ float red[256];
  const int row = blockIdx.x;
  const int t   = threadIdx.x;
  const float* xr = x + (size_t)row * ncols;
  float s = 0.0f;
  for (int i = t; i < ncols; i += 256) { float v = xr[i]; s += v * v; }
  red[t] = s; __syncthreads();
  for (int off = 128; off > 0; off >>= 1) { if (t < off) red[t] += red[t + off]; __syncthreads(); }
  const float inv = 1.0f / (sqrtf(red[0] / (float)ncols) + 1e-6f);
  for (int i = t; i < ncols; i += 256)
    out[(size_t)row * ncols + i] = f2bf_u(xr[i] * inv * w[i]);
}

// silu(V) -> bf16; RoPE(Q),RoPE(K) -> bf16
__global__ void uvqk_post_kernel(const float* __restrict__ uvqk, const float* __restrict__ pe,
                                 unsigned short* __restrict__ vout,
                                 unsigned short* __restrict__ qout,
                                 unsigned short* __restrict__ kout) {
  const int row = blockIdx.x;                 // 0..4095
  const int l   = row % LL;
  const float* base  = uvqk + (size_t)row * UVQK;
  const float* perow = pe + (size_t)l * KK;
  for (int i = threadIdx.x; i < EE; i += 256) {
    float v = base[EE + i];
    vout[(size_t)row * EE + i] = f2bf_u(v / (1.0f + expf(-v)));
  }
  const int t = threadIdx.x;
  const int u = t & 127;
  const float* src = base + ((t < 128) ? (2 * EE) : (2 * EE + KK));
  const float sinv = perow[u & 63], cosv = perow[64 + (u & 63)];
  const float x1 = src[u & 63], x2 = src[64 + (u & 63)];
  const float r = (u < 64) ? (x1 * cosv - x2 * sinv) : (x2 * cosv + x1 * sinv);
  unsigned short* dst = (t < 128) ? qout : kout;
  dst[(size_t)row * KK + u] = f2bf_u(r);
}

__global__ void softmax_kernel(const float* __restrict__ s, unsigned short* __restrict__ attn) {
  __shared__ float red[256];
  const int row = blockIdx.x;
  const int t   = threadIdx.x;
  const float v = s[(size_t)row * LL + t] * 0.088388347648318447f;  // K^-0.5
  red[t] = v; __syncthreads();
  for (int off = 128; off > 0; off >>= 1) { if (t < off) red[t] = fmaxf(red[t], red[t + off]); __syncthreads(); }
  const float m = red[0]; __syncthreads();
  const float e = expf(v - m);
  red[t] = e; __syncthreads();
  for (int off = 128; off > 0; off >>= 1) { if (t < off) red[t] += red[t + off]; __syncthreads(); }
  attn[(size_t)row * LL + t] = f2bf_u(e / red[0]);
}

__global__ void gate_mul_kernel(const float* __restrict__ uvqk, const float* __restrict__ av,
                                unsigned short* __restrict__ o) {
  const size_t idx = (size_t)blockIdx.x * 256 + threadIdx.x;  // ROWS*E
  if (idx >= (size_t)ROWS * EE) return;
  const size_t row = idx / EE;
  const int    col = (int)(idx % EE);
  const float  uv  = uvqk[row * UVQK + col];
  o[idx] = f2bf_u((uv / (1.0f + expf(-uv))) * av[idx]);
}

__global__ void unpatch_kernel(const float* __restrict__ hout, float* __restrict__ out) {
  const int idx = blockIdx.x * 256 + threadIdx.x;    // B*3*128*128
  if (idx >= BB * 3 * 128 * 128) return;
  const int xx = idx & 127;
  const int yy = (idx >> 7) & 127;
  const int c  = (idx >> 14) % 3;
  const int b  = idx / (3 * 128 * 128);
  const int gy = yy >> 3, py = yy & 7, gx = xx >> 3, px = xx & 7;
  const int l  = gy * 16 + gx;
  const int pd = (py * 8 + px) * 3 + c;
  out[idx] = hout[((size_t)b * LL + l) * PD + pd];
}

// ---------------------------------------------------------------------------
// Launcher
// ---------------------------------------------------------------------------
extern "C" void kernel_launch(void* const* d_in, const int* in_sizes, int n_in,
                              void* d_out, int out_size, void* d_ws, size_t ws_size,
                              hipStream_t stream) {
  (void)in_sizes; (void)n_in; (void)out_size; (void)ws_size;

  const float* x         = (const float*)d_in[0];
  const int*   t_idx     = (const int*)d_in[1];
  const float* patch_W   = (const float*)d_in[2];
  const float* t_emb     = (const float*)d_in[3];
  const float* Wuv       = (const float*)d_in[4];
  const float* Wout      = (const float*)d_in[5];
  const float* gnorm     = (const float*)d_in[6];
  const float* fnorm_w   = (const float*)d_in[7];
  const float* unpatch_W = (const float*)d_in[8];
  float* out = (float*)d_out;

  char* p = (char*)d_ws;
  auto alloc = [&](size_t bytes) -> char* {
    char* q = p; p += (bytes + 255) & ~(size_t)255; return q;
  };
  float*          pe     = (float*)         alloc((size_t)LL * KK * 4);
  unsigned short* xp     = (unsigned short*)alloc((size_t)ROWS * PD * 2);
  float*          h      = (float*)         alloc((size_t)ROWS * HH * 4);
  unsigned short* xn     = (unsigned short*)alloc((size_t)ROWS * HH * 2);
  float*          uvqk   = (float*)         alloc((size_t)ROWS * UVQK * 4);
  unsigned short* vbuf   = (unsigned short*)alloc((size_t)ROWS * EE * 2);
  unsigned short* qbuf   = (unsigned short*)alloc((size_t)ROWS * KK * 2);
  unsigned short* kbuf   = (unsigned short*)alloc((size_t)ROWS * KK * 2);
  float*          scores = (float*)         alloc((size_t)ROWS * LL * 4);
  unsigned short* attn   = (unsigned short*)alloc((size_t)ROWS * LL * 2);
  float*          av     = (float*)         alloc((size_t)ROWS * EE * 4);
  unsigned short* obuf   = (unsigned short*)alloc((size_t)ROWS * EE * 2);
  float*          hout   = (float*)         alloc((size_t)ROWS * PD * 4);

  pe_kernel<<<LL, 128, 0, stream>>>(pe);
  patchify_kernel<<<(BB * LL * PD + 255) / 256, 256, 0, stream>>>(x, xp);

  // h = xp @ patch_W  (4096x192x768)
  gemm_bf16_wmma<0, 0><<<dim3(HH / 128, ROWS / 128, 1), 256, 0, stream>>>(
      xp, patch_W, h, ROWS, HH, PD, PD, HH, HH, 0, 0, 0);
  add_temb_kernel<<<(ROWS * HH + 255) / 256, 256, 0, stream>>>(h, t_emb, t_idx);

  for (int layer = 0; layer < 24; ++layer) {
    rmsnorm_kernel<<<ROWS, 256, 0, stream>>>(h, gnorm + (size_t)layer * HH, xn, HH);
    // uvqk = xn @ Wuv[layer]  (4096x768x3328)
    gemm_bf16_wmma<0, 0><<<dim3(UVQK / 128, ROWS / 128, 1), 256, 0, stream>>>(
        xn, Wuv + (size_t)layer * HH * UVQK, uvqk, ROWS, UVQK, HH, HH, UVQK, UVQK, 0, 0, 0);
    uvqk_post_kernel<<<ROWS, 256, 0, stream>>>(uvqk, pe, vbuf, qbuf, kbuf);
    // scores[b] = q[b] @ k[b]^T  (256x128x256, batched)
    gemm_bf16_wmma<2, 0><<<dim3(2, 2, BB), 256, 0, stream>>>(
        qbuf, kbuf, scores, LL, LL, KK, KK, KK, LL,
        (long long)LL * KK, (long long)LL * KK, (long long)LL * LL);
    softmax_kernel<<<ROWS, 256, 0, stream>>>(scores, attn);
    // av[b] = attn[b] @ v[b]  (256x256x1536, batched)
    gemm_bf16_wmma<1, 0><<<dim3(EE / 128, 2, BB), 256, 0, stream>>>(
        attn, vbuf, av, LL, EE, LL, LL, EE, EE,
        (long long)LL * LL, (long long)LL * EE, (long long)LL * EE);
    gate_mul_kernel<<<(int)(((size_t)ROWS * EE + 255) / 256), 256, 0, stream>>>(uvqk, av, obuf);
    // h += o @ Wout[layer]  (4096x1536x768, residual epilogue)
    gemm_bf16_wmma<0, 1><<<dim3(HH / 128, ROWS / 128, 1), 256, 0, stream>>>(
        obuf, Wout + (size_t)layer * EE * HH, h, ROWS, HH, EE, EE, HH, HH, 0, 0, 0);
  }

  rmsnorm_kernel<<<ROWS, 256, 0, stream>>>(h, fnorm_w, xn, HH);
  // hout = xn @ unpatch_W  (4096x768x192, ragged N)
  gemm_bf16_wmma<0, 0><<<dim3((PD + 127) / 128, ROWS / 128, 1), 256, 0, stream>>>(
      xn, unpatch_W, hout, ROWS, PD, HH, HH, PD, PD, 0, 0, 0);
  unpatch_kernel<<<(BB * 3 * 128 * 128 + 255) / 256, 256, 0, stream>>>(hout, out);
}
